// AffinityNeuralNetworkCliffNetF_29300266893466
// MI455X (gfx1250) — matrix operand order repaired and checked by reading, not compiled, not run on hardware
//
#include <hip/hip_runtime.h>
#include <hip/hip_bf16.h>

typedef _Float16 f16;
typedef __attribute__((ext_vector_type(16))) _Float16 v16h;
typedef __attribute__((ext_vector_type(8)))  _Float16 v8h_t;
typedef __attribute__((ext_vector_type(8)))  float    v8f;

#define LEAKY_SLOPE 0.1f
#define NEGV -1e30f

__device__ __forceinline__ float leaky_f(float x) { return x > 0.f ? x : LEAKY_SLOPE * x; }

__device__ __forceinline__ void atomicMaxF32(float* addr, float val) {
    unsigned int* ua = (unsigned int*)addr;
    unsigned int old = *ua;
    while (true) {
        float f = __uint_as_float(old);
        if (f >= val) break;
        unsigned int assumed = old;
        old = atomicCAS(ua, assumed, __float_as_uint(val));
        if (old == assumed) break;
    }
}

// ---- A-fragment loader (16x32 f16), ISA 7.12.2 16-bit A layout ----
// lane L: row m=L&15 ; koff=(L>>4)*8 ; slots 0..7 -> K=koff+0..7 ; slots 8..15 -> K=koff+16..23
__device__ __forceinline__ v16h load_a_f16(const f16* __restrict__ p) {
    v8h_t lo = *reinterpret_cast<const v8h_t*>(p);
    v8h_t hi = *reinterpret_cast<const v8h_t*>(p + 16);
    v16h a;
#pragma unroll
    for (int s = 0; s < 8; ++s) { a[s] = lo[s]; a[8 + s] = hi[s]; }
    return a;
}

__device__ __forceinline__ v16h load_a_f32(const float* __restrict__ p) {
    float4 x0 = *reinterpret_cast<const float4*>(p);
    float4 x1 = *reinterpret_cast<const float4*>(p + 4);
    float4 y0 = *reinterpret_cast<const float4*>(p + 16);
    float4 y1 = *reinterpret_cast<const float4*>(p + 20);
    v16h a;
    a[0]=(f16)x0.x; a[1]=(f16)x0.y; a[2]=(f16)x0.z; a[3]=(f16)x0.w;
    a[4]=(f16)x1.x; a[5]=(f16)x1.y; a[6]=(f16)x1.z; a[7]=(f16)x1.w;
    a[8]=(f16)y0.x; a[9]=(f16)y0.y; a[10]=(f16)y0.z; a[11]=(f16)y0.w;
    a[12]=(f16)y1.x; a[13]=(f16)y1.y; a[14]=(f16)y1.z; a[15]=(f16)y1.w;
    return a;
}

// ---- B-fragment from pre-swizzled f16 weights: one 32B contiguous load per lane ----
// Wz layout: [((kt*16 + nt)*32 + lane)*16 + s] = W[kt*32 + (lane>>4)*8 + (s&7) + (s>>3)*16][nt*16 + (lane&15)]
__device__ __forceinline__ v16h load_b_swz(const f16* __restrict__ Wz, int kt, int nt, int lane) {
    return *reinterpret_cast<const v16h*>(Wz + ((((size_t)kt * 16 + nt) * 32 + lane) << 4));
}

// ---- weight swizzle prep: W[K x 256] f32 -> Wz f16 in B-fragment order ----
__global__ void swz_weight(const float* __restrict__ W, f16* __restrict__ Wz, int K)
{
    int i = blockIdx.x * blockDim.x + threadIdx.x;
    if (i >= K * 256) return;
    int s    = i & 15;
    int lane = (i >> 4) & 31;
    int nt   = (i >> 9) & 15;
    int kt   = i >> 13;
    int k = kt * 32 + ((lane >> 4) << 3) + (s & 7) + ((s >> 3) << 4);
    int n = (nt << 4) + (lane & 15);
    Wz[i] = (f16)W[(size_t)k * 256 + n];
}

// =====================================================================
// GEMM: out[M x 256] = epilogue(A[M x K] @ W[K x 256] + bias)
// grid.x = M/32 ; 128 threads = 4 waves ; wave w -> col tiles [w*4, w*4+4)
// each wave: 2 M-tiles x 4 N-tiles = 8 WMMA per k-step ; K compile-time,
// fully unrolled so the scheduler can run loads deep ahead of the WMMAs.
// EPI: 0 = store f16 ; 1 = atomicAdd pool[seg[row]] ; 2 = atomicMax pool[seg[row]]
// =====================================================================
template<int EPI, bool LEAKY, bool A_F16, int K>
__global__ __launch_bounds__(128)
void gemm256(const void* __restrict__ Aptr, const f16* __restrict__ Wz,
             const float* __restrict__ bias, f16* __restrict__ outF16,
             float* __restrict__ pool, const int* __restrict__ seg)
{
    const int lane = threadIdx.x & 31;
    const int wave = threadIdx.x >> 5;
    const int row0 = blockIdx.x * 32;
    const int m    = lane & 15;
    const int koff = (lane >> 4) * 8;
    const int nt0  = wave * 4;

    v8f acc[2][4] = {};
#pragma unroll
    for (int kt = 0; kt < (K >> 5); ++kt) {
        const int k0 = kt << 5;
        v16h a0, a1;
        if (A_F16) {
            const f16* Af = (const f16*)Aptr;
            a0 = load_a_f16(Af + (size_t)(row0 + m)      * K + k0 + koff);
            a1 = load_a_f16(Af + (size_t)(row0 + 16 + m) * K + k0 + koff);
        } else {
            const float* Af = (const float*)Aptr;
            a0 = load_a_f32(Af + (size_t)(row0 + m)      * K + k0 + koff);
            a1 = load_a_f32(Af + (size_t)(row0 + 16 + m) * K + k0 + koff);
        }
#pragma unroll
        for (int t = 0; t < 4; ++t) {
            v16h b = load_b_swz(Wz, kt, nt0 + t, lane);
            acc[0][t] = __builtin_amdgcn_wmma_f32_16x16x32_f16(
                false, a0, false, b, (short)0, acc[0][t], false, false);
            acc[1][t] = __builtin_amdgcn_wmma_f32_16x16x32_f16(
                false, a1, false, b, (short)0, acc[1][t], false, false);
        }
    }
    // C layout: lane L, VGPR r -> row = rbase + r, col = (nt0+t)*16 + (L&15)
#pragma unroll
    for (int u = 0; u < 2; ++u) {
        const int rbase = row0 + u * 16 + (lane >> 4) * 8;
#pragma unroll
        for (int t = 0; t < 4; ++t) {
            const int col = (nt0 + t) * 16 + (lane & 15);
            const float bv = bias[col];
#pragma unroll
            for (int r = 0; r < 8; ++r) {
                float v = acc[u][t][r] + bv;
                if (LEAKY) v = leaky_f(v);
                const int row = rbase + r;
                if (EPI == 0) {
                    outF16[(size_t)row * 256 + col] = (f16)v;
                } else {
                    float* dst = pool + (size_t)seg[row] * 256 + col;
                    if (EPI == 1) atomicAdd(dst, v);
                    else          atomicMaxF32(dst, v);
                }
            }
        }
    }
}

// =====================================================================
// feat GEMM: out = leaky([comp_emb | supe[bc] | pool[bc]] @ W[768x256] + bias)
// K=768 peeled into three fixed-pointer segments so every A load is a
// plain global_load_b128 (no v_cndmask pointer selects / FLAT path).
// =====================================================================
__global__ __launch_bounds__(128)
void gemm_feat(const f16* __restrict__ compEmb, const f16* __restrict__ supe,
               const f16* __restrict__ pool, const int* __restrict__ bc,
               const f16* __restrict__ Wz, const float* __restrict__ bias,
               f16* __restrict__ outF16)
{
    const int lane = threadIdx.x & 31;
    const int wave = threadIdx.x >> 5;
    const int row0 = blockIdx.x * 32;
    const int m    = lane & 15;
    const int koff = (lane >> 4) * 8;
    const int nt0  = wave * 4;

    const int rowA = row0 + m;
    const int rowB = row0 + 16 + m;
    const int segA = bc[rowA];
    const int segB = bc[rowB];

    v8f acc[2][4] = {};

    auto do_seg = [&](const f16* __restrict__ sA, const f16* __restrict__ sB, int ktbase) {
#pragma unroll
        for (int kk = 0; kk < 8; ++kk) {
            const int kt = ktbase + kk;
            v16h a0 = load_a_f16(sA + (kk << 5) + koff);
            v16h a1 = load_a_f16(sB + (kk << 5) + koff);
#pragma unroll
            for (int t = 0; t < 4; ++t) {
                v16h b = load_b_swz(Wz, kt, nt0 + t, lane);
                acc[0][t] = __builtin_amdgcn_wmma_f32_16x16x32_f16(
                    false, a0, false, b, (short)0, acc[0][t], false, false);
                acc[1][t] = __builtin_amdgcn_wmma_f32_16x16x32_f16(
                    false, a1, false, b, (short)0, acc[1][t], false, false);
            }
        }
    };

    do_seg(compEmb + (size_t)rowA * 256, compEmb + (size_t)rowB * 256, 0);
    do_seg(supe    + (size_t)segA * 256, supe    + (size_t)segB * 256, 8);
    do_seg(pool    + (size_t)segA * 256, pool    + (size_t)segB * 256, 16);

#pragma unroll
    for (int u = 0; u < 2; ++u) {
        const int rbase = row0 + u * 16 + (lane >> 4) * 8;
#pragma unroll
        for (int t = 0; t < 4; ++t) {
            const int col = (nt0 + t) * 16 + (lane & 15);
            const float bv = bias[col];
#pragma unroll
            for (int r = 0; r < 8; ++r) {
                float v = leaky_f(acc[u][t][r] + bv);
                outF16[(size_t)(rbase + r) * 256 + col] = (f16)v;
            }
        }
    }
}

// ---- GEMV: out[row] = dot(H[row,0:256], w) + wb[0] ; one wave per row ----
__global__ __launch_bounds__(256)
void rowdot(const f16* __restrict__ Hm, const float* __restrict__ w,
            const float* __restrict__ wb, float* __restrict__ out, int nrows)
{
    const int lane = threadIdx.x & 31;
    const int row = blockIdx.x * 8 + (threadIdx.x >> 5);
    if (row >= nrows) return;
    const f16* p = Hm + (size_t)row * 256;
    float s = 0.f;
#pragma unroll
    for (int j = lane; j < 256; j += 32) s += (float)p[j] * w[j];
#pragma unroll
    for (int off = 16; off; off >>= 1) s += __shfl_down(s, off, 32);
    if (lane == 0) out[row] = s + wb[0];
}

// ---- workspace init ----
__global__ void init_ws(float* supe, float* pool, float* segmax, float* segsum, int nBH, int nB)
{
    int i = blockIdx.x * blockDim.x + threadIdx.x;
    if (i < nBH) { supe[i] = 0.f; pool[i] = NEGV; }
    if (i < nB)  { segmax[i] = NEGV; segsum[i] = 0.f; }
}

__global__ void cvt_pools(const float* __restrict__ a, const float* __restrict__ b,
                          f16* __restrict__ oa, f16* __restrict__ ob, int n)
{
    int i = blockIdx.x * blockDim.x + threadIdx.x;
    if (i < n) { oa[i] = (f16)a[i]; ob[i] = (f16)b[i]; }
}

__global__ void seg_max_scalar(const float* __restrict__ pre, const int* __restrict__ bc,
                               float* __restrict__ segmax, int n)
{
    int i = blockIdx.x * blockDim.x + threadIdx.x;
    if (i < n) atomicMaxF32(&segmax[bc[i]], pre[i]);
}

__global__ void seg_exp(float* __restrict__ pre, const int* __restrict__ bc,
                        const float* __restrict__ segmax, float* __restrict__ segsum, int n)
{
    int i = blockIdx.x * blockDim.x + threadIdx.x;
    if (i < n) {
        float e = expf(pre[i] - segmax[bc[i]]);
        pre[i] = e;
        atomicAdd(&segsum[bc[i]], e);
    }
}

__global__ void seg_alpha(const float* __restrict__ aexp, const int* __restrict__ bc,
                          const float* __restrict__ segsum, float* __restrict__ alpha, int n)
{
    int i = blockIdx.x * blockDim.x + threadIdx.x;
    if (i < n) alpha[i] = aexp[i] / (segsum[bc[i]] + 1e-6f);
}

// ---- deterministic per-segment weighted sum (batch_comp sorted) ----
__global__ __launch_bounds__(256)
void seg_vector(const f16* __restrict__ raw, const float* __restrict__ alpha,
                const int* __restrict__ bc, float* __restrict__ vec, int n)
{
    __shared__ int sh[2];
    const int b = blockIdx.x;
    if (threadIdx.x < 2) {
        int target = b + (int)threadIdx.x;
        int lo = 0, hi = n;
        while (lo < hi) { int mid = (lo + hi) >> 1; if (bc[mid] < target) lo = mid + 1; else hi = mid; }
        sh[threadIdx.x] = lo;
    }
    __syncthreads();
    const int lo = sh[0], hi = sh[1];
    const int c = threadIdx.x;
    float s = 0.f;
    for (int i = lo; i < hi; ++i) s += alpha[i] * (float)raw[(size_t)i * 256 + c];
    vec[(size_t)b * 256 + c] = s;
}

extern "C" void kernel_launch(void* const* d_in, const int* in_sizes, int n_in,
                              void* d_out, int out_size, void* d_ws, size_t ws_size,
                              hipStream_t stream)
{
    const float* comp_feature = (const float*)d_in[0];
    const float* prot_feature = (const float*)d_in[1];
    const int*   batch_comp   = (const int*)d_in[2];
    const int*   batch_prot   = (const int*)d_in[3];
    const float* c_aff_W = (const float*)d_in[4];   const float* c_aff_b = (const float*)d_in[5];
    const float* c_sup_W = (const float*)d_in[6];   const float* c_sup_b = (const float*)d_in[7];
    const float* p_aff_W = (const float*)d_in[8];   const float* p_aff_b = (const float*)d_in[9];
    const float* raw_W1  = (const float*)d_in[10];  const float* raw_b1  = (const float*)d_in[11];
    const float* raw_W2  = (const float*)d_in[12];  const float* raw_b2  = (const float*)d_in[13];
    const float* alpha_W1= (const float*)d_in[14];  const float* alpha_b1= (const float*)d_in[15];
    const float* alpha_W2= (const float*)d_in[16];  const float* alpha_b2= (const float*)d_in[17];
    const float* out_W1  = (const float*)d_in[18];  const float* out_b1  = (const float*)d_in[19];
    const float* out_W2  = (const float*)d_in[20];  const float* out_b2  = (const float*)d_in[21];

    const int N = in_sizes[2];   // 262144
    const int B = 4096, H = 256;

    // workspace layout (~420 MB)
    char* w = (char*)d_ws;
    f16* comp_emb = (f16*)w;  w += (size_t)N * H * sizeof(f16);
    f16* hbuf     = (f16*)w;  w += (size_t)N * H * sizeof(f16);
    f16* rawbuf   = (f16*)w;  w += (size_t)N * H * sizeof(f16);
    float* supe32 = (float*)w; w += (size_t)B * H * sizeof(float);
    float* pool32 = (float*)w; w += (size_t)B * H * sizeof(float);
    f16* supe16   = (f16*)w;  w += (size_t)B * H * sizeof(f16);
    f16* pool16   = (f16*)w;  w += (size_t)B * H * sizeof(f16);
    float* pre    = (float*)w; w += (size_t)N * sizeof(float);
    float* segmax = (float*)w; w += (size_t)B * sizeof(float);
    float* segsum = (float*)w; w += (size_t)B * sizeof(float);
    // pre-swizzled f16 weights (B-fragment layout)
    f16* c_aff_Wz = (f16*)w;  w += (size_t)256 * 256 * sizeof(f16);
    f16* c_sup_Wz = (f16*)w;  w += (size_t)256 * 256 * sizeof(f16);
    f16* p_aff_Wz = (f16*)w;  w += (size_t)256 * 256 * sizeof(f16);
    f16* raw_W1z  = (f16*)w;  w += (size_t)768 * 256 * sizeof(f16);
    f16* alpha_W1z= (f16*)w;  w += (size_t)768 * 256 * sizeof(f16);
    f16* raw_W2z  = (f16*)w;  w += (size_t)256 * 256 * sizeof(f16);
    f16* out_W1z  = (f16*)w;  w += (size_t)256 * 256 * sizeof(f16);

    float* out_vec   = (float*)d_out;            // [B,256]
    float* out_alpha = out_vec + (size_t)B * H;  // [N]
    float* out_aff   = out_alpha + (size_t)N;    // [B]

    const dim3 blk(128);
    const int rowTiles = N / 32;

    init_ws<<<(B * H + 255) / 256, 256, 0, stream>>>(supe32, pool32, segmax, segsum, B * H, B);

    // one-off weight swizzles (tiny)
    swz_weight<<<(256 * 256 + 255) / 256, 256, 0, stream>>>(c_aff_W, c_aff_Wz, 256);
    swz_weight<<<(256 * 256 + 255) / 256, 256, 0, stream>>>(c_sup_W, c_sup_Wz, 256);
    swz_weight<<<(256 * 256 + 255) / 256, 256, 0, stream>>>(p_aff_W, p_aff_Wz, 256);
    swz_weight<<<(768 * 256 + 255) / 256, 256, 0, stream>>>(raw_W1, raw_W1z, 768);
    swz_weight<<<(768 * 256 + 255) / 256, 256, 0, stream>>>(alpha_W1, alpha_W1z, 768);
    swz_weight<<<(256 * 256 + 255) / 256, 256, 0, stream>>>(raw_W2, raw_W2z, 256);
    swz_weight<<<(256 * 256 + 255) / 256, 256, 0, stream>>>(out_W1, out_W1z, 256);

    // comp_emb = leaky(comp @ c_aff_W + b)            (stored f16)
    gemm256<0, true, false, 256><<<rowTiles, blk, 0, stream>>>(
        comp_feature, c_aff_Wz, c_aff_b, comp_emb, nullptr, nullptr);
    // supe[b] += leaky(comp @ c_sup_W + b)            (segment-sum epilogue)
    gemm256<1, true, false, 256><<<rowTiles, blk, 0, stream>>>(
        comp_feature, c_sup_Wz, c_sup_b, nullptr, supe32, batch_comp);
    // pool[b] = max(leaky(prot @ p_aff_W + b))        (segment-max epilogue)
    gemm256<2, true, false, 256><<<rowTiles, blk, 0, stream>>>(
        prot_feature, p_aff_Wz, p_aff_b, nullptr, pool32, batch_prot);

    cvt_pools<<<(B * H + 255) / 256, 256, 0, stream>>>(supe32, pool32, supe16, pool16, B * H);

    // alpha path: h = leaky(feat @ alpha_W1 + b1) ; prealpha = h @ alpha_W2 + b2
    gemm_feat<<<rowTiles, blk, 0, stream>>>(comp_emb, supe16, pool16, batch_comp,
                                            alpha_W1z, alpha_b1, hbuf);
    rowdot<<<(N + 7) / 8, 256, 0, stream>>>(hbuf, alpha_W2, alpha_b2, pre, N);

    // scatter softmax over graphs
    seg_max_scalar<<<(N + 255) / 256, 256, 0, stream>>>(pre, batch_comp, segmax, N);
    seg_exp<<<(N + 255) / 256, 256, 0, stream>>>(pre, batch_comp, segmax, segsum, N);
    seg_alpha<<<(N + 255) / 256, 256, 0, stream>>>(pre, batch_comp, segsum, out_alpha, N);

    // raw path: h = leaky(feat @ raw_W1 + b1) ; raw = h @ raw_W2 + b2
    gemm_feat<<<rowTiles, blk, 0, stream>>>(comp_emb, supe16, pool16, batch_comp,
                                            raw_W1z, raw_b1, hbuf);
    gemm256<0, false, true, 256><<<rowTiles, blk, 0, stream>>>(
        hbuf, raw_W2z, raw_b2, rawbuf, nullptr, nullptr);

    // vector = segment_sum(raw * alpha)  (deterministic, sorted segments)
    seg_vector<<<B, 256, 0, stream>>>(rawbuf, out_alpha, batch_comp, out_vec, N);

    // affinity = leaky(vector @ out_W1 + b1) @ out_W2 + b2
    gemm256<0, true, false, 256><<<B / 32, blk, 0, stream>>>(
        out_vec, out_W1z, out_b1, hbuf, nullptr, nullptr);
    rowdot<<<(B + 7) / 8, 256, 0, stream>>>(hbuf, out_W2, out_b2, out_aff, B);
}
